// TransformerVAE_84387517432108
// MI455X (gfx1250) — matrix operand order repaired and checked by reading, not compiled, not run on hardware
//
#include <hip/hip_runtime.h>
#include <hip/hip_bf16.h>
#include <math.h>

// ---------------------------------------------------------------------------
// Problem constants (from reference)
// ---------------------------------------------------------------------------
constexpr int kB   = 32;
constexpr int kS   = 512;
constexpr int kV   = 8000;
constexpr int kE   = 512;
constexpr int kLAT = 512;
constexpr int kD   = kE + kLAT;   // 1024 decoder width
constexpr int kH   = 4;           // heads
constexpr int kFF  = 2048;
constexpr float kNEG = -1.0e9f;

typedef __attribute__((ext_vector_type(16))) __bf16 v16bf;
typedef __attribute__((ext_vector_type(8)))  __bf16 v8bf;
typedef __attribute__((ext_vector_type(8)))  float  v8f;

__device__ inline v8f zero8() {
  v8f z = {0.f, 0.f, 0.f, 0.f, 0.f, 0.f, 0.f, 0.f};
  return z;
}
__device__ inline __bf16 f2bf(float f) { return (__bf16)f; }
__device__ inline v16bf cat8(v8bf lo, v8bf hi) {
  return __builtin_shufflevector(lo, hi, 0, 1, 2, 3, 4, 5, 6, 7, 8, 9, 10, 11,
                                 12, 13, 14, 15);
}

// 16-byte global -> LDS async copy (gfx1250 GLOBAL_LOAD_ASYNC_TO_LDS_B128,
// tracked by ASYNCcnt).  lds_off = LDS byte offset (low 32 bits of the
// shared-aperture generic address, ISA 10.2).
__device__ inline void async_cp16(unsigned lds_off, const void* gaddr) {
  asm volatile("global_load_async_to_lds_b128 %0, %1, off" ::"v"(lds_off),
               "v"(gaddr)
               : "memory");
}
__device__ inline void wait_async_le6() {
  asm volatile("s_wait_asynccnt 0x6" ::: "memory");
}
__device__ inline void wait_async_0() {
  asm volatile("s_wait_asynccnt 0x0" ::: "memory");
}

// ---------------------------------------------------------------------------
// Register-blocked bf16 GEMM with double-buffered async-LDS staging:
//   D[M,N] = act(A[M,K]bf16 @ W[N,K]^T bf16 + bias[N] (+ res[M,N]f32))
// Block = 4 waves (128 thr) on a 128x64 macro-tile.  Per 32-deep k-step the
// block async-copies A(128x32) + B(64x32) into LDS (next buffer) while the
// current buffer feeds 8 v_wmma per wave.  LDS rows padded to 80B.
// ---------------------------------------------------------------------------
template <int ACT, bool HAS_RES, bool MASK_ROWS, bool OUT_BF16>
__global__ __launch_bounds__(128) void gemm_bf16(
    const __bf16* __restrict__ A, const __bf16* __restrict__ W,
    const float* __restrict__ bias, const float* __restrict__ res,
    float* __restrict__ Df, __bf16* __restrict__ Db, int M, int N, int K,
    const int* __restrict__ lengths) {
  constexpr int LDT = 40;  // padded LDS row stride in elements (80 bytes)
  __shared__ __align__(16) __bf16 smA[2][128 * LDT];
  __shared__ __align__(16) __bf16 smB[2][64 * LDT];

  const int tid  = threadIdx.x;
  const int wave = tid >> 5;
  const int lane = tid & 31;
  const int g    = lane >> 4;
  const int l16  = lane & 15;
  const int mBlk = blockIdx.y * 128;
  const int nBlk = blockIdx.x * 64;

  // Issue one k-step tile group: 4 A-chunks + 2 B-chunks of 16B per thread
  // -> 6 ASYNCcnt increments per wave per tile group.
  auto issueTile = [&](int buf, int k0) {
#pragma unroll
    for (int i = 0; i < 4; ++i) {
      const int c = tid + i * 128;      // 512 chunks cover 128x32 A tile
      const int row = c >> 2, seg = c & 3;
      const __bf16* src = A + (size_t)(mBlk + row) * K + k0 + seg * 8;
      async_cp16((unsigned)(uintptr_t)&smA[buf][row * LDT + seg * 8], src);
    }
#pragma unroll
    for (int i = 0; i < 2; ++i) {
      const int c = tid + i * 128;      // 256 chunks cover 64x32 B tile
      const int row = c >> 2, seg = c & 3;
      const __bf16* src = W + (size_t)(nBlk + row) * K + k0 + seg * 8;
      async_cp16((unsigned)(uintptr_t)&smB[buf][row * LDT + seg * 8], src);
    }
  };

  v8f acc[2][4];
#pragma unroll
  for (int mt = 0; mt < 2; ++mt)
#pragma unroll
    for (int nt = 0; nt < 4; ++nt) acc[mt][nt] = zero8();

  issueTile(0, 0);

  int buf = 0;
  for (int k0 = 0; k0 < K; k0 += 32, buf ^= 1) {
    const bool more = (k0 + 32 < K);
    if (more) issueTile(buf ^ 1, k0 + 32);
    // ASYNCcnt completes in order: waiting <=6 retires the current tile
    // group while the 6 just-issued next-tile copies stream in.
    if (more)
      wait_async_le6();
    else
      wait_async_0();
    __syncthreads();

    const __bf16* As = smA[buf];
    const __bf16* Bs = smB[buf];
    v16bf af[2], bfr[4];
#pragma unroll
    for (int mt = 0; mt < 2; ++mt) {
      const __bf16* ar = As + (wave * 32 + mt * 16 + l16) * LDT + g * 8;
      af[mt] = cat8(*(const v8bf*)(ar), *(const v8bf*)(ar + 16));
    }
#pragma unroll
    for (int nt = 0; nt < 4; ++nt) {
      const __bf16* br = Bs + (nt * 16 + l16) * LDT + g * 8;
      bfr[nt] = cat8(*(const v8bf*)(br), *(const v8bf*)(br + 16));
    }
#pragma unroll
    for (int mt = 0; mt < 2; ++mt)
#pragma unroll
      for (int nt = 0; nt < 4; ++nt)
        acc[mt][nt] = __builtin_amdgcn_wmma_f32_16x16x32_bf16(
            false, af[mt], false, bfr[nt], (short)0, acc[mt][nt], false, false);
    __syncthreads();  // all waves done reading before buf is overwritten
  }

#pragma unroll
  for (int nt = 0; nt < 4; ++nt) {
    const int n    = nBlk + nt * 16 + l16;
    const float bv = bias[n];
#pragma unroll
    for (int mt = 0; mt < 2; ++mt) {
#pragma unroll
      for (int r = 0; r < 8; ++r) {
        const int m = mBlk + wave * 32 + mt * 16 + g * 8 + r;
        float v = acc[mt][nt][r] + bv;
        if constexpr (HAS_RES) v += res[(size_t)m * N + n];
        if constexpr (ACT == 1) v = fmaxf(v, 0.0f);
        if constexpr (MASK_ROWS) {
          const int b = m >> 9, s = m & (kS - 1);
          if (s >= lengths[b]) v = 0.0f;
        }
        if constexpr (OUT_BF16)
          Db[(size_t)m * N + n] = f2bf(v);
        else
          Df[(size_t)m * N + n] = v;
      }
    }
  }
}

// ---------------------------------------------------------------------------
// Small f32 GEMM (M=32 VAE-head projections): wave per 16x16 tile, cvt inline.
// ---------------------------------------------------------------------------
__global__ __launch_bounds__(32) void gemm_small(
    const float* __restrict__ A, const float* __restrict__ W,
    const float* __restrict__ bias, float* __restrict__ D, int M, int N,
    int K) {
  const int tn = blockIdx.x, tm = blockIdx.y;
  const int lane = threadIdx.x & 31;
  const int g = lane >> 4, l16 = lane & 15;
  const float* arow = A + (size_t)(tm * 16 + l16) * K;
  const float* wrow = W + (size_t)(tn * 16 + l16) * K + g * 16;
  v8f acc = zero8();
  for (int k0 = 0; k0 < K; k0 += 32) {
    v16bf av, bv;
#pragma unroll
    for (int e = 0; e < 8; ++e) {
      av[e]     = f2bf(arow[k0 + g * 16 + e]);
      av[e + 8] = f2bf(arow[k0 + g * 16 + 8 + e]);
    }
#pragma unroll
    for (int e = 0; e < 16; ++e) bv[e] = f2bf(wrow[k0 + e]);
    acc = __builtin_amdgcn_wmma_f32_16x16x32_bf16(false, av, false, bv,
                                                  (short)0, acc, false, false);
  }
  const int n = tn * 16 + l16;
  const float bv = bias[n];
#pragma unroll
  for (int r = 0; r < 8; ++r) {
    const int m = tm * 16 + g * 8 + r;
    D[(size_t)m * N + n] = acc[r] + bv;
  }
}

// ---------------------------------------------------------------------------
// Flash attention over bf16 QKV.  One wave per (b, h, 16-query tile, split).
// S^T = K·Q^T (lane owns a query column), online softmax in registers,
// out^T = V^T · P^T.  Writes bf16 attention output.
// ---------------------------------------------------------------------------
template <int DH, int NT_OUT, bool CAUSAL>
__global__ __launch_bounds__(32) void flash_attn(
    const __bf16* __restrict__ QKV, __bf16* __restrict__ O,
    const int* __restrict__ lengths, int dmodel) {
  constexpr int KC = DH / 32;
  const int bh     = blockIdx.x;
  const int b      = bh >> 2;
  const int h      = bh & (kH - 1);
  const int qt     = blockIdx.y;
  const int dhBase = blockIdx.z * (NT_OUT * 16);
  const int lane   = threadIdx.x & 31;
  const int g      = lane >> 4;
  const int q      = lane & 15;
  const int qglob  = qt * 16 + q;

  const size_t strideTok = (size_t)3 * dmodel;
  const __bf16* base  = QKV + (size_t)b * kS * strideTok;
  const __bf16* Qrow  = base + (size_t)qglob * strideTok + h * DH;
  const __bf16* Kbase = base + dmodel + h * DH;
  const __bf16* Vbase = base + 2 * dmodel + h * DH;

  v16bf qb[KC];  // Q^T as WMMA B operand (loop invariant)
#pragma unroll
  for (int kc = 0; kc < KC; ++kc)
    qb[kc] = *(const v16bf*)(Qrow + kc * 32 + g * 16);

  v8f oacc[NT_OUT];
#pragma unroll
  for (int t = 0; t < NT_OUT; ++t) oacc[t] = zero8();

  float mrun = -3.0e38f, lrun = 0.0f;
  int lenb = 0;
  if constexpr (!CAUSAL) lenb = lengths[b];
  const float scale = (DH == 128) ? 0.0883883476483184f : 0.0625f;

  for (int kb = 0; kb < kS / 32; ++kb) {
    if constexpr (CAUSAL) {
      if (kb * 32 > qt * 16 + 15) break;
    }
    float pf[2][8];
#pragma unroll
    for (int t = 0; t < 2; ++t) {
      v8f c = zero8();
      const __bf16* krow = Kbase + (size_t)(kb * 32 + t * 16 + q) * strideTok;
#pragma unroll
      for (int kc = 0; kc < KC; ++kc) {
        v16bf av = cat8(*(const v8bf*)(krow + kc * 32 + g * 16),
                        *(const v8bf*)(krow + kc * 32 + g * 16 + 8));
        c = __builtin_amdgcn_wmma_f32_16x16x32_bf16(false, av, false, qb[kc],
                                                    (short)0, c, false, false);
      }
#pragma unroll
      for (int r = 0; r < 8; ++r) {
        const int key = kb * 32 + t * 16 + g * 8 + r;
        float sv = c[r] * scale;
        if constexpr (CAUSAL) {
          if (key > qglob) sv = kNEG;
        } else {
          if (key >= lenb) sv = kNEG;
        }
        pf[t][r] = sv;
      }
    }
    float mx = -3.0e38f;
#pragma unroll
    for (int t = 0; t < 2; ++t)
#pragma unroll
      for (int r = 0; r < 8; ++r) mx = fmaxf(mx, pf[t][r]);
    mx = fmaxf(mx, __shfl_xor(mx, 16, 32));
    const float mnew  = fmaxf(mrun, mx);
    const float alpha = __expf(mrun - mnew);
    float lsum = 0.0f;
#pragma unroll
    for (int t = 0; t < 2; ++t)
#pragma unroll
      for (int r = 0; r < 8; ++r) {
        const float p = __expf(pf[t][r] - mnew);
        pf[t][r] = p;
        lsum += p;
      }
    lsum += __shfl_xor(lsum, 16, 32);
    lrun = lrun * alpha + lsum;
    mrun = mnew;
#pragma unroll
    for (int nt = 0; nt < NT_OUT; ++nt)
#pragma unroll
      for (int r = 0; r < 8; ++r) oacc[nt][r] *= alpha;

    float ps[2][8];
#pragma unroll
    for (int t = 0; t < 2; ++t)
#pragma unroll
      for (int r = 0; r < 8; ++r) ps[t][r] = __shfl_xor(pf[t][r], 16, 32);
    v16bf pb;
#pragma unroll
    for (int e = 0; e < 8; ++e) {
      pb[e]     = f2bf((g == 0) ? pf[0][e] : ps[1][e]);
      pb[e + 8] = f2bf((g == 0) ? ps[0][e] : pf[1][e]);
    }
#pragma unroll
    for (int nt = 0; nt < NT_OUT; ++nt) {
      const int dhc = dhBase + nt * 16 + q;
      v16bf av;
#pragma unroll
      for (int e = 0; e < 8; ++e) {
        av[e] = Vbase[(size_t)(kb * 32 + g * 16 + e) * strideTok + dhc];
        av[e + 8] =
            Vbase[(size_t)(kb * 32 + g * 16 + 8 + e) * strideTok + dhc];
      }
      oacc[nt] = __builtin_amdgcn_wmma_f32_16x16x32_bf16(
          false, av, false, pb, (short)0, oacc[nt], false, false);
    }
  }

  const float inv = 1.0f / lrun;
  __bf16* orow = O + ((size_t)b * kS + qglob) * dmodel + h * DH + dhBase;
#pragma unroll
  for (int nt = 0; nt < NT_OUT; ++nt)
#pragma unroll
    for (int r = 0; r < 8; ++r)
      orow[nt * 16 + g * 8 + r] = f2bf(oacc[nt][r] * inv);
}

// ---------------------------------------------------------------------------
// Elementwise / reduction helpers
// ---------------------------------------------------------------------------
__global__ void cvt_bf16_kernel(const float* __restrict__ src,
                                __bf16* __restrict__ dst, size_t n) {
  size_t i = (size_t)blockIdx.x * blockDim.x + threadIdx.x;
  if (i < n) dst[i] = f2bf(src[i]);
}

__global__ void embed_kernel(const int* __restrict__ ids,
                             const float* __restrict__ emb,
                             float* __restrict__ outf,
                             __bf16* __restrict__ outb) {
  size_t i = (size_t)blockIdx.x * blockDim.x + threadIdx.x;
  if (i >= (size_t)kB * kS * kE) return;
  const int e = (int)(i % kE);
  const size_t tok = i / kE;
  const float v = emb[(size_t)ids[tok] * kE + e];
  outf[i] = v;
  outb[i] = f2bf(v);
}

__global__ void ln_kernel(const float* __restrict__ X,
                          const float* __restrict__ gam,
                          const float* __restrict__ bet,
                          float* __restrict__ Yf, __bf16* __restrict__ Yb,
                          int d) {
  __shared__ float s1[256], s2[256];
  const int row = blockIdx.x;
  const float* x = X + (size_t)row * d;
  float sum = 0.f, sq = 0.f;
  for (int i = threadIdx.x; i < d; i += 256) {
    const float v = x[i];
    sum += v;
    sq += v * v;
  }
  s1[threadIdx.x] = sum;
  s2[threadIdx.x] = sq;
  __syncthreads();
  for (int s = 128; s > 0; s >>= 1) {
    if ((int)threadIdx.x < s) {
      s1[threadIdx.x] += s1[threadIdx.x + s];
      s2[threadIdx.x] += s2[threadIdx.x + s];
    }
    __syncthreads();
  }
  const float mean = s1[0] / d;
  const float var  = s2[0] / d - mean * mean;
  const float inv  = rsqrtf(var + 1e-5f);
  for (int i = threadIdx.x; i < d; i += 256) {
    const float y = (x[i] - mean) * inv * gam[i] + bet[i];
    Yf[(size_t)row * d + i] = y;
    Yb[(size_t)row * d + i] = f2bf(y);
  }
}

__global__ void pool_kernel(const float* __restrict__ X, float* __restrict__ P) {
  const int i = blockIdx.x * blockDim.x + threadIdx.x;
  if (i >= kB * kE) return;
  const int b = i / kE, e = i % kE;
  float s = 0.f;
  for (int t = 0; t < kS; ++t) s += X[((size_t)b * kS + t) * kE + e];
  P[i] = s * (1.0f / kS);
}

__global__ void reparam_kernel(const float* __restrict__ mu,
                               const float* __restrict__ logvar,
                               const float* __restrict__ eps,
                               float* __restrict__ z) {
  const int i = blockIdx.x * blockDim.x + threadIdx.x;
  if (i >= kB * kLAT) return;
  z[i] = mu[i] + eps[i] * __expf(0.5f * logvar[i]);
}

__global__ void decin_kernel(const int* __restrict__ ids,
                             const float* __restrict__ emb,
                             const float* __restrict__ start,
                             const float* __restrict__ zc,
                             float* __restrict__ outf,
                             __bf16* __restrict__ outb) {
  size_t i = (size_t)blockIdx.x * blockDim.x + threadIdx.x;
  if (i >= (size_t)kB * kS * kD) return;
  const int c = (int)(i % kD);
  const size_t bs = i / kD;
  const int s = (int)(bs % kS);
  const int b = (int)(bs / kS);
  float v;
  if (c < kE) {
    v = (s == 0) ? start[c] : emb[(size_t)ids[b * kS + s - 1] * kE + c];
  } else {
    v = zc[(size_t)b * kLAT + (c - kE)];
  }
  outf[i] = v;
  outb[i] = f2bf(v);
}

__global__ void maskout_kernel(const int* __restrict__ lengths,
                               float* __restrict__ mout) {
  const int i = blockIdx.x * blockDim.x + threadIdx.x;
  if (i >= kB * kS) return;
  const int b = i >> 9, s = i & (kS - 1);
  mout[i] = (s < lengths[b]) ? 1.0f : 0.0f;
}

// ---------------------------------------------------------------------------
// Host-side orchestration
// ---------------------------------------------------------------------------
static inline void cvt(const float* src, __bf16* dst, size_t n,
                       hipStream_t stream) {
  cvt_bf16_kernel<<<(unsigned)((n + 255) / 256), 256, 0, stream>>>(src, dst, n);
}

extern "C" void kernel_launch(void* const* d_in, const int* in_sizes, int n_in,
                              void* d_out, int out_size, void* d_ws,
                              size_t ws_size, hipStream_t stream) {
  (void)in_sizes; (void)n_in; (void)ws_size;
  const int*   token_ids = (const int*)d_in[0];
  const int*   lengths   = (const int*)d_in[1];
  const float* eps       = (const float*)d_in[2];
  const float* token_emb = (const float*)d_in[3];
  const float* start_tok = (const float*)d_in[4];
  const float* fc_mu_w   = (const float*)d_in[5];
  const float* fc_mu_b   = (const float*)d_in[6];
  const float* fc_lv_w   = (const float*)d_in[7];
  const float* fc_lv_b   = (const float*)d_in[8];
  const float* fc_de_w   = (const float*)d_in[9];
  const float* fc_de_b   = (const float*)d_in[10];
  const float* out_w     = (const float*)d_in[11];
  const float* out_b     = (const float*)d_in[12];
  const float* e_qkv_w = (const float*)d_in[13];
  const float* e_qkv_b = (const float*)d_in[14];
  const float* e_ow    = (const float*)d_in[15];
  const float* e_ob    = (const float*)d_in[16];
  const float* e_ln1g  = (const float*)d_in[17];
  const float* e_ln1b  = (const float*)d_in[18];
  const float* e_ln2g  = (const float*)d_in[19];
  const float* e_ln2b  = (const float*)d_in[20];
  const float* e_w1    = (const float*)d_in[21];
  const float* e_b1    = (const float*)d_in[22];
  const float* e_w2    = (const float*)d_in[23];
  const float* e_b2    = (const float*)d_in[24];
  const float* d_qkv_w = (const float*)d_in[25];
  const float* d_qkv_b = (const float*)d_in[26];
  const float* d_ow    = (const float*)d_in[27];
  const float* d_ob    = (const float*)d_in[28];
  const float* d_ln1g  = (const float*)d_in[29];
  const float* d_ln1b  = (const float*)d_in[30];
  const float* d_ln2g  = (const float*)d_in[31];
  const float* d_ln2b  = (const float*)d_in[32];
  const float* d_w1    = (const float*)d_in[33];
  const float* d_b1    = (const float*)d_in[34];
  const float* d_w2    = (const float*)d_in[35];
  const float* d_b2    = (const float*)d_in[36];

  const int BS = kB * kS;  // 16384 rows

  // ---- workspace layout (256B aligned chunks) -----------------------------
  char* wsb = (char*)d_ws;
  size_t off = 0;
  auto alloc = [&](size_t bytes) {
    void* p = wsb + off;
    off += (bytes + 255) & ~(size_t)255;
    return p;
  };
  float*  xbuf   = (float*)alloc((size_t)BS * kD * 4);       // residual (f32)
  float*  tbuf   = (float*)alloc((size_t)BS * kD * 4);       // pre-LN (f32)
  __bf16* xbf    = (__bf16*)alloc((size_t)BS * kD * 2);      // GEMM A (bf16)
  __bf16* qkvbf  = (__bf16*)alloc((size_t)BS * 3 * kD * 2);  // qkv / ff hidden
  __bf16* attnbf = (__bf16*)alloc((size_t)BS * kD * 2);      // attention out
  __bf16* ffbf   = qkvbf;                                    // alias (ff hidden)
  __bf16* eqkvwB = (__bf16*)alloc((size_t)2 * 3 * kE * kE * 2);
  __bf16* eowB   = (__bf16*)alloc((size_t)2 * kE * kE * 2);
  __bf16* ew1B   = (__bf16*)alloc((size_t)2 * kFF * kE * 2);
  __bf16* ew2B   = (__bf16*)alloc((size_t)2 * kE * kFF * 2);
  __bf16* dqkvwB = (__bf16*)alloc((size_t)2 * 3 * kD * kD * 2);
  __bf16* dowB   = (__bf16*)alloc((size_t)2 * kD * kD * 2);
  __bf16* dw1B   = (__bf16*)alloc((size_t)2 * kFF * kD * 2);
  __bf16* dw2B   = (__bf16*)alloc((size_t)2 * kD * kFF * 2);
  __bf16* outwB  = (__bf16*)alloc((size_t)kV * kD * 2);
  float* pooled = (float*)alloc((size_t)kB * kE * 4);
  float* mu     = (float*)alloc((size_t)kB * kLAT * 4);
  float* logvar = (float*)alloc((size_t)kB * kLAT * 4);
  float* zbuf   = (float*)alloc((size_t)kB * kLAT * 4);
  float* zcbuf  = (float*)alloc((size_t)kB * kLAT * 4);

  // ---- convert weights to bf16 once per launch ----------------------------
  cvt(e_qkv_w, eqkvwB, (size_t)2 * 3 * kE * kE, stream);
  cvt(e_ow,    eowB,   (size_t)2 * kE * kE, stream);
  cvt(e_w1,    ew1B,   (size_t)2 * kFF * kE, stream);
  cvt(e_w2,    ew2B,   (size_t)2 * kE * kFF, stream);
  cvt(d_qkv_w, dqkvwB, (size_t)2 * 3 * kD * kD, stream);
  cvt(d_ow,    dowB,   (size_t)2 * kD * kD, stream);
  cvt(d_w1,    dw1B,   (size_t)2 * kFF * kD, stream);
  cvt(d_w2,    dw2B,   (size_t)2 * kD * kFF, stream);
  cvt(out_w,   outwB,  (size_t)kV * kD, stream);

  // ---- token embeddings -> encoder input ----------------------------------
  {
    const size_t tot = (size_t)BS * kE;
    embed_kernel<<<(unsigned)((tot + 255) / 256), 256, 0, stream>>>(
        token_ids, token_emb, xbuf, xbf);
  }

  // ---- encoder stack (2 layers, d = 512) ----------------------------------
  for (int i = 0; i < 2; ++i) {
    gemm_bf16<0, false, false, true>
        <<<dim3(3 * kE / 64, BS / 128), 128, 0, stream>>>(
            xbf, eqkvwB + (size_t)i * 3 * kE * kE,
            e_qkv_b + (size_t)i * 3 * kE, nullptr, nullptr, qkvbf, BS, 3 * kE,
            kE, nullptr);
    flash_attn<128, 8, false><<<dim3(kB * kH, kS / 16, 1), 32, 0, stream>>>(
        qkvbf, attnbf, lengths, kE);
    gemm_bf16<0, true, false, false>
        <<<dim3(kE / 64, BS / 128), 128, 0, stream>>>(
            attnbf, eowB + (size_t)i * kE * kE, e_ob + (size_t)i * kE, xbuf,
            tbuf, nullptr, BS, kE, kE, nullptr);
    ln_kernel<<<BS, 256, 0, stream>>>(tbuf, e_ln1g + (size_t)i * kE,
                                      e_ln1b + (size_t)i * kE, xbuf, xbf, kE);
    gemm_bf16<1, false, false, true>
        <<<dim3(kFF / 64, BS / 128), 128, 0, stream>>>(
            xbf, ew1B + (size_t)i * kFF * kE, e_b1 + (size_t)i * kFF, nullptr,
            nullptr, ffbf, BS, kFF, kE, nullptr);
    gemm_bf16<0, true, false, false>
        <<<dim3(kE / 64, BS / 128), 128, 0, stream>>>(
            ffbf, ew2B + (size_t)i * kE * kFF, e_b2 + (size_t)i * kE, xbuf,
            tbuf, nullptr, BS, kE, kFF, nullptr);
    ln_kernel<<<BS, 256, 0, stream>>>(tbuf, e_ln2g + (size_t)i * kE,
                                      e_ln2b + (size_t)i * kE, xbuf, xbf, kE);
  }

  // ---- VAE head -----------------------------------------------------------
  pool_kernel<<<(kB * kE + 255) / 256, 256, 0, stream>>>(xbuf, pooled);
  gemm_small<<<dim3(kLAT / 16, kB / 16), 32, 0, stream>>>(pooled, fc_mu_w,
                                                          fc_mu_b, mu, kB,
                                                          kLAT, kE);
  gemm_small<<<dim3(kLAT / 16, kB / 16), 32, 0, stream>>>(pooled, fc_lv_w,
                                                          fc_lv_b, logvar, kB,
                                                          kLAT, kE);
  reparam_kernel<<<(kB * kLAT + 255) / 256, 256, 0, stream>>>(mu, logvar, eps,
                                                              zbuf);
  gemm_small<<<dim3(kLAT / 16, kB / 16), 32, 0, stream>>>(zbuf, fc_de_w,
                                                          fc_de_b, zcbuf, kB,
                                                          kLAT, kLAT);

  // ---- decoder input [B,S,E+LAT] ------------------------------------------
  {
    const size_t tot = (size_t)BS * kD;
    decin_kernel<<<(unsigned)((tot + 255) / 256), 256, 0, stream>>>(
        token_ids, token_emb, start_tok, zcbuf, xbuf, xbf);
  }

  // ---- decoder stack (2 layers, d = 1024, causal) -------------------------
  for (int i = 0; i < 2; ++i) {
    gemm_bf16<0, false, false, true>
        <<<dim3(3 * kD / 64, BS / 128), 128, 0, stream>>>(
            xbf, dqkvwB + (size_t)i * 3 * kD * kD,
            d_qkv_b + (size_t)i * 3 * kD, nullptr, nullptr, qkvbf, BS, 3 * kD,
            kD, nullptr);
    flash_attn<256, 8, true><<<dim3(kB * kH, kS / 16, 2), 32, 0, stream>>>(
        qkvbf, attnbf, nullptr, kD);
    gemm_bf16<0, true, false, false>
        <<<dim3(kD / 64, BS / 128), 128, 0, stream>>>(
            attnbf, dowB + (size_t)i * kD * kD, d_ob + (size_t)i * kD, xbuf,
            tbuf, nullptr, BS, kD, kD, nullptr);
    ln_kernel<<<BS, 256, 0, stream>>>(tbuf, d_ln1g + (size_t)i * kD,
                                      d_ln1b + (size_t)i * kD, xbuf, xbf, kD);
    gemm_bf16<1, false, false, true>
        <<<dim3(kFF / 64, BS / 128), 128, 0, stream>>>(
            xbf, dw1B + (size_t)i * kFF * kD, d_b1 + (size_t)i * kFF, nullptr,
            nullptr, ffbf, BS, kFF, kD, nullptr);
    gemm_bf16<0, true, false, false>
        <<<dim3(kD / 64, BS / 128), 128, 0, stream>>>(
            ffbf, dw2B + (size_t)i * kD * kFF, d_b2 + (size_t)i * kD, xbuf,
            tbuf, nullptr, BS, kD, kFF, nullptr);
    ln_kernel<<<BS, 256, 0, stream>>>(tbuf, d_ln2g + (size_t)i * kD,
                                      d_ln2b + (size_t)i * kD, xbuf, xbf, kD);
  }

  // ---- logits projection with fused row masking ---------------------------
  float* logits = (float*)d_out;
  gemm_bf16<0, false, true, false><<<dim3(kV / 64, BS / 128), 128, 0, stream>>>(
      xbf, outwB, out_b, nullptr, logits, nullptr, BS, kV, kD, lengths);

  // ---- second output: mask [B,S] as 0/1 floats ----------------------------
  if (out_size >= kB * kS * kV + kB * kS) {
    maskout_kernel<<<(kB * kS + 255) / 256, 256, 0, stream>>>(
        lengths, logits + (size_t)kB * kS * kV);
  }
}
